// Codebook_24635932410208
// MI455X (gfx1250) — matrix operand order repaired and checked by reading, not compiled, not run on hardware
//
#include <hip/hip_runtime.h>
#include <hip/hip_bf16.h>

typedef __attribute__((ext_vector_type(16))) _Float16 v16h;
typedef __attribute__((ext_vector_type(8)))  float    v8f;

#define DECAY_F 0.8f
#define OMD_F   0.2f
#define EPS_F   1e-5f
#define NTOK    65536
#define CODES   2048
#define DIMD    64

// ---------------- workspace layout (bytes) ----------------
// emb_h : _Float16[CODES*DIMD]  [0,        262144)
// e2    : float[CODES]          [262144,   270336)
// counts: float[CODES]          [270336,   278528)
// esum  : float[CODES*DIMD]     [278528,   802816)

// ---------------------------------------------------------------------------
// Prep: convert embeddings fp32 -> f16, compute e2[c] = sum(emb[c]^2),
// zero the scatter accumulators.
// ---------------------------------------------------------------------------
__global__ void vq_prep(const float* __restrict__ emb,
                        _Float16* __restrict__ emb_h,
                        float* __restrict__ e2,
                        float* __restrict__ counts,
                        float* __restrict__ esum) {
    int tid = blockIdx.x * blockDim.x + threadIdx.x;   // 131072 threads
    if (tid < CODES * DIMD) esum[tid] = 0.0f;
    if (tid < CODES) {
        counts[tid] = 0.0f;
        const float* row = emb + (size_t)tid * DIMD;
        _Float16* dst = emb_h + (size_t)tid * DIMD;
        float s = 0.0f;
#pragma unroll
        for (int j = 0; j < DIMD; ++j) {
            float v = row[j];
            s += v * v;
            dst[j] = (_Float16)v;
        }
        e2[tid] = s;
    }
}

// ---------------------------------------------------------------------------
// Main: WMMA distance GEMM + argmin + gather + scatter.
// 256 threads = 8 waves; each wave handles 32 tokens (2 M-tiles of 16).
// Embeddings staged through LDS in 256-code tiles shared by all 8 waves.
// ---------------------------------------------------------------------------
__global__ __launch_bounds__(256)
void vq_main(const float* __restrict__ x,
             const float* __restrict__ emb,
             const _Float16* __restrict__ emb_h,
             const float* __restrict__ e2g,
             float* __restrict__ counts,
             float* __restrict__ esum,
             float* __restrict__ out_q,
             float* __restrict__ out_ind) {
    __shared__ __align__(64) _Float16 sEmb[256 * DIMD];  // 32 KB
    __shared__ float sE2[256];

    const int tid  = threadIdx.x;
    const int lane = tid & 31;
    const int wave = tid >> 5;
    const int n    = lane & 15;   // column / token-row selector
    const int hi   = lane >> 4;   // half-wave selector
    const int base = blockIdx.x * 256 + wave * 32;   // first token of this wave

    // ---- A fragments: 2 m-tiles x 2 K-halves, per ISA 16-bit A layout ----
    // lane (m = n, hi): halves e<8 -> k = kb + 8*hi + e ; e>=8 -> k = kb + 16 + 8*hi + (e-8)
    v16h a[2][2];
#pragma unroll
    for (int t = 0; t < 2; ++t) {
        const float* xrow = x + (size_t)(base + t * 16 + n) * DIMD;
#pragma unroll
        for (int kk = 0; kk < 2; ++kk) {
            const int kb = kk * 32;
            const float4* p0 = (const float4*)(xrow + kb + 8 * hi);
            const float4* p1 = (const float4*)(xrow + kb + 16 + 8 * hi);
            float4 f0 = p0[0], f1 = p0[1], f2 = p1[0], f3 = p1[1];
            v16h av;
            av[0]  = (_Float16)f0.x; av[1]  = (_Float16)f0.y; av[2]  = (_Float16)f0.z; av[3]  = (_Float16)f0.w;
            av[4]  = (_Float16)f1.x; av[5]  = (_Float16)f1.y; av[6]  = (_Float16)f1.z; av[7]  = (_Float16)f1.w;
            av[8]  = (_Float16)f2.x; av[9]  = (_Float16)f2.y; av[10] = (_Float16)f2.z; av[11] = (_Float16)f2.w;
            av[12] = (_Float16)f3.x; av[13] = (_Float16)f3.y; av[14] = (_Float16)f3.z; av[15] = (_Float16)f3.w;
            a[t][kk] = av;
        }
    }

    float best[2][8];
    int   bidx[2][8];
#pragma unroll
    for (int t = 0; t < 2; ++t)
#pragma unroll
        for (int r = 0; r < 8; ++r) { best[t][r] = 3.4e38f; bidx[t][r] = 0; }

    for (int pass = 0; pass < CODES / 256; ++pass) {
        __syncthreads();
        {   // cooperative LDS fill: thread tid copies one code row (128 B)
            const uint4* src = (const uint4*)(emb_h + (size_t)(pass * 256 + tid) * DIMD);
            uint4* dst = (uint4*)(sEmb + tid * DIMD);
#pragma unroll
            for (int j = 0; j < 8; ++j) dst[j] = src[j];
            sE2[tid] = e2g[pass * 256 + tid];
        }
        __syncthreads();

        for (int sub = 0; sub < 16; ++sub) {
            const int clb = sub * 16;
            // B fragment (32x16, K-major per lane): lane column n, e<8 -> k = 8*hi + e ; e>=8 -> k = 16 + 8*hi + (e-8)
            const _Float16* brow = sEmb + (clb + n) * DIMD;
            v16h b00 = *(const v16h*)(brow), b01 = *(const v16h*)(brow + 16);
            v16h b10 = *(const v16h*)(brow + 32), b11 = *(const v16h*)(brow + 48);
            v16h blo = hi ? __builtin_shufflevector(b00, b01, 8,9,10,11,12,13,14,15,24,25,26,27,28,29,30,31)
                          : __builtin_shufflevector(b00, b01, 0,1,2,3,4,5,6,7,16,17,18,19,20,21,22,23);   // K = 0..31
            v16h bhi = hi ? __builtin_shufflevector(b10, b11, 8,9,10,11,12,13,14,15,24,25,26,27,28,29,30,31)
                          : __builtin_shufflevector(b10, b11, 0,1,2,3,4,5,6,7,16,17,18,19,20,21,22,23);   // K = 32..63
            const float e2v = sE2[clb + n];
            const int   c   = pass * 256 + clb + n;
#pragma unroll
            for (int t = 0; t < 2; ++t) {
                v8f acc = {};
                acc = __builtin_amdgcn_wmma_f32_16x16x32_f16(
                    false, a[t][0], false, blo, (short)0, acc, false, false);
                acc = __builtin_amdgcn_wmma_f32_16x16x32_f16(
                    false, a[t][1], false, bhi, (short)0, acc, false, false);
#pragma unroll
                for (int r = 0; r < 8; ++r) {
                    float s = e2v - 2.0f * acc[r];       // == |x-e|^2 - |x|^2 (monotone)
                    if (s < best[t][r]) { best[t][r] = s; bidx[t][r] = c; }
                }
            }
        }
    }

    // ---- butterfly argmin across the 16 lanes sharing each token row ----
#pragma unroll
    for (int t = 0; t < 2; ++t)
#pragma unroll
        for (int r = 0; r < 8; ++r)
#pragma unroll
            for (int m = 1; m < 16; m <<= 1) {
                float os = __shfl_xor(best[t][r], m, 32);
                int   oi = __shfl_xor(bidx[t][r], m, 32);
                if (os < best[t][r] || (os == best[t][r] && oi < bidx[t][r])) {
                    best[t][r] = os; bidx[t][r] = oi;
                }
            }

    // ---- outputs: indices, gathered quantize rows, scatter accumulators ----
#pragma unroll
    for (int t = 0; t < 2; ++t) {
#pragma unroll
        for (int r = 0; r < 8; ++r) {
            const int token = base + t * 16 + r + 8 * hi;  // D row m = r + 8*hi
            const int idx   = bidx[t][r];
            if (n == 0) {
                out_ind[token] = (float)idx;
                atomicAdd(&counts[idx], 1.0f);
            }
#pragma unroll
            for (int j = 0; j < 4; ++j) {
                const int dcol = n + 16 * j;
                out_q[(size_t)token * DIMD + dcol] = emb[(size_t)idx * DIMD + dcol];
                atomicAdd(&esum[(size_t)idx * DIMD + dcol],
                          x[(size_t)token * DIMD + dcol]);
            }
        }
    }
}

// ---------------------------------------------------------------------------
// Finalize: EMA update, Laplace smoothing, renormalized embeddings.
// ---------------------------------------------------------------------------
__global__ void vq_finalize(const float* __restrict__ cs,
                            const float* __restrict__ ea,
                            const float* __restrict__ counts,
                            const float* __restrict__ esum,
                            float* __restrict__ out_ne,
                            float* __restrict__ out_ncs,
                            float* __restrict__ out_nea) {
    __shared__ float sNcs[CODES];
    __shared__ float sRed[1024];
    const int tid = threadIdx.x;   // one block of 1024

    float local = 0.0f;
    for (int i = tid; i < CODES; i += 1024) {
        float v = cs[i] * DECAY_F + counts[i] * OMD_F;
        sNcs[i] = v;
        out_ncs[i] = v;
        local += v;
    }
    sRed[tid] = local;
    __syncthreads();
    for (int s = 512; s > 0; s >>= 1) {
        if (tid < s) sRed[tid] += sRed[tid + s];
        __syncthreads();
    }
    const float tot   = sRed[0];
    const float denom = tot + (float)CODES * EPS_F;

    for (int i = tid; i < CODES * DIMD; i += 1024) {
        const int c = i >> 6;
        float nea = ea[i] * DECAY_F + esum[i] * OMD_F;
        out_nea[i] = nea;
        float sm = (sNcs[c] + EPS_F) / denom * tot;
        out_ne[i] = nea / sm;
    }
}

// ---------------------------------------------------------------------------
extern "C" void kernel_launch(void* const* d_in, const int* in_sizes, int n_in,
                              void* d_out, int out_size, void* d_ws, size_t ws_size,
                              hipStream_t stream) {
    const float* x   = (const float*)d_in[0];   // (16,4096,64)
    const float* emb = (const float*)d_in[1];   // (1,2048,64)
    const float* cs  = (const float*)d_in[2];   // (1,2048)
    const float* ea  = (const float*)d_in[3];   // (1,2048,64)
    float* out = (float*)d_out;

    char* ws = (char*)d_ws;
    _Float16* emb_h = (_Float16*)(ws);
    float* e2     = (float*)(ws + 262144);
    float* counts = (float*)(ws + 270336);
    float* esum   = (float*)(ws + 278528);

    float* out_q   = out;               // 4,194,304  quantize
    float* out_ind = out + 4194304;     //    65,536  embed_ind (as float)
    float* out_ne  = out + 4259840;     //   131,072  new_embeddings
    float* out_ncs = out + 4390912;     //     2,048  new_cluster_size
    float* out_nea = out + 4392960;     //   131,072  new_embed_avg

    vq_prep<<<512, 256, 0, stream>>>(emb, emb_h, e2, counts, esum);
    vq_main<<<NTOK / 256, 256, 0, stream>>>(x, emb, emb_h, e2, counts, esum,
                                            out_q, out_ind);
    vq_finalize<<<1, 1024, 0, stream>>>(cs, ea, counts, esum,
                                        out_ne, out_ncs, out_nea);
}
